// MultiHeadAttention_1340029796531
// MI455X (gfx1250) — matrix operand order repaired and checked by reading, not compiled, not run on hardware
//
#include <hip/hip_runtime.h>
#include <hip/hip_bf16.h>
#include <cstdint>

// ---- problem constants (hardcoded from reference) ----
#define B_DIM  2
#define S_DIM  2048
#define D_DIM  1024
#define H_DIM  16
#define HD_DIM 64
#define M_DIM  (B_DIM * S_DIM)      // 4096 rows of x

typedef __bf16 bf16_t;
typedef __attribute__((ext_vector_type(16))) __bf16 v16bf;
typedef __attribute__((ext_vector_type(8)))  __bf16 v8bf;
typedef __attribute__((ext_vector_type(8)))  float  v8f;

union V16 { v16bf v; v8bf h[2]; };

__device__ __forceinline__ v8bf load8bf(const bf16_t* p) {
  return *(const v8bf*)p;
}

// --- gfx1250 async global->LDS path (guarded; fallback = direct loads) ---
#if defined(__has_builtin)
#if __has_builtin(__builtin_amdgcn_global_load_async_to_lds_b128) && \
    __has_builtin(__builtin_amdgcn_s_wait_asynccnt)
#define USE_ASYNC_LDS 1
#endif
#endif

#ifdef USE_ASYNC_LDS
// builtin signature is (v4i addrspace(1)*, v4i addrspace(3)*, imm, imm)
typedef int tile128 __attribute__((vector_size(16)));
#define AS1P(p) ((__attribute__((address_space(1))) tile128*)(uintptr_t)(p))
#define AS3P(p) ((__attribute__((address_space(3))) tile128*)(p))
#endif

// ---------------------------------------------------------------------------
// f32 -> bf16 conversion (grid-stride)
// ---------------------------------------------------------------------------
__global__ void cvt_f32_bf16_kernel(const float* __restrict__ in,
                                    bf16_t* __restrict__ out, int n) {
  int i = blockIdx.x * blockDim.x + threadIdx.x;
  int stride = gridDim.x * blockDim.x;
  for (; i < n; i += stride) out[i] = (bf16_t)in[i];
}

// ---------------------------------------------------------------------------
// GEMM: C[M,N] = A[M,K](bf16) @ W[N,K]^T(bf16) + bias
// mode 0: store bf16 row-major [M,N]
// mode 1: store bf16 "V-transposed": m -> (b, s); out[b*D*S + n*S + s]
// mode 2: store f32 row-major [M,N]
// Block: 256 thr = 8 waves (2x4), wave tile 32x64, block tile 64x256, BK=32.
// Async variant: A(64x32) + B(256x32) tiles double-buffered in LDS via
// global_load_async_to_lds_b128 (5 per thread per k-chunk), waited with
// s_wait_asynccnt; rows padded to 40 elems (80B) -> conflict-free ds_load_b128.
// ---------------------------------------------------------------------------
__global__ __launch_bounds__(256)
void gemm_xwT_kernel(const bf16_t* __restrict__ A, const bf16_t* __restrict__ W,
                     const float* __restrict__ bias, void* __restrict__ Cout,
                     int M, int N, int K, int mode) {
  const int tid  = threadIdx.x;
  const int lane = tid & 31;
  const int wid  = tid >> 5;     // 0..7
  const int wm   = wid >> 2;     // 0..1
  const int wn   = wid & 3;      // 0..3
  const int lrow = lane & 15;
  const int lhi  = lane >> 4;    // 0 or 1

  const int mblk = blockIdx.x * 64;
  const int nblk = blockIdx.y * 256;
  const int mbase = mblk + wm * 32;
  const int nbase = nblk + wn * 64;

  v8f acc[2][4] = {};

#ifdef USE_ASYNC_LDS
  __shared__ __align__(16) bf16_t Abuf[2][64][40];    // 64x32 tile, padded rows
  __shared__ __align__(16) bf16_t Bbuf[2][256][40];   // 256x32 tile, padded rows

  const int crow = tid >> 2;        // 0..63
  const int cseg = (tid & 3) * 8;   // element offset 0,8,16,24 (16B segments)

  auto issue_tile = [&](int buf, int k) {
    const bf16_t* as = A + (size_t)(mblk + crow) * K + k + cseg;
    __builtin_amdgcn_global_load_async_to_lds_b128(AS1P(as),
        AS3P(&Abuf[buf][crow][cseg]), 0, 0);
#pragma unroll
    for (int i = 0; i < 4; ++i) {
      const int brow = crow + i * 64;
      const bf16_t* bs = W + (size_t)(nblk + brow) * K + k + cseg;
      __builtin_amdgcn_global_load_async_to_lds_b128(AS1P(bs),
          AS3P(&Bbuf[buf][brow][cseg]), 0, 0);
    }
  };

  issue_tile(0, 0);
  for (int k = 0; k < K; k += 32) {
    const int cur = (k >> 5) & 1;
    if (k + 32 < K) {
      issue_tile(cur ^ 1, k + 32);
      __builtin_amdgcn_s_wait_asynccnt(5);   // current buffer's 5 ops done
    } else {
      __builtin_amdgcn_s_wait_asynccnt(0);
    }
    __syncthreads();                         // tile visible to all waves

    V16 a[2], bm[4];
#pragma unroll
    for (int mt = 0; mt < 2; ++mt) {
      const bf16_t* ar = &Abuf[cur][wm * 32 + mt * 16 + lrow][lhi * 8];
      a[mt].h[0] = *(const v8bf*)ar;
      a[mt].h[1] = *(const v8bf*)(ar + 16);
    }
#pragma unroll
    for (int nt = 0; nt < 4; ++nt) {
      const bf16_t* br = &Bbuf[cur][wn * 64 + nt * 16 + lrow][lhi * 8];
      bm[nt].h[0] = *(const v8bf*)br;
      bm[nt].h[1] = *(const v8bf*)(br + 16);
    }
#pragma unroll
    for (int mt = 0; mt < 2; ++mt)
#pragma unroll
      for (int nt = 0; nt < 4; ++nt)
        acc[mt][nt] = __builtin_amdgcn_wmma_f32_16x16x32_bf16(
            false, a[mt].v, false, bm[nt].v, (short)0, acc[mt][nt], false, false);

    __syncthreads();   // all reads done before buffer is overwritten
  }
#else
  for (int k = 0; k < K; k += 32) {
    V16 a[2], bm[4];
#pragma unroll
    for (int mt = 0; mt < 2; ++mt) {
      const bf16_t* ar = A + ((size_t)(mbase + mt * 16 + lrow)) * K + k + lhi * 8;
      a[mt].h[0] = load8bf(ar);
      a[mt].h[1] = load8bf(ar + 16);
      __builtin_prefetch(ar + 32, 0, 1);
    }
#pragma unroll
    for (int nt = 0; nt < 4; ++nt) {
      const bf16_t* wr = W + ((size_t)(nbase + nt * 16 + lrow)) * K + k + lhi * 8;
      bm[nt].h[0] = load8bf(wr);
      bm[nt].h[1] = load8bf(wr + 16);
      __builtin_prefetch(wr + 32, 0, 1);
    }
#pragma unroll
    for (int mt = 0; mt < 2; ++mt)
#pragma unroll
      for (int nt = 0; nt < 4; ++nt)
        acc[mt][nt] = __builtin_amdgcn_wmma_f32_16x16x32_bf16(
            false, a[mt].v, false, bm[nt].v, (short)0, acc[mt][nt], false, false);
  }
#endif

#pragma unroll
  for (int nt = 0; nt < 4; ++nt) {
    const int n = nbase + nt * 16 + lrow;
    const float bv = bias ? bias[n] : 0.0f;
#pragma unroll
    for (int mt = 0; mt < 2; ++mt)
#pragma unroll
      for (int r = 0; r < 8; ++r) {
        const int m = mbase + mt * 16 + r + 8 * lhi;   // C-layout: M=r+8*(lane>>4)
        const float v = acc[mt][nt][r] + bv;
        if (mode == 0) {
          ((bf16_t*)Cout)[(size_t)m * N + n] = (bf16_t)v;
        } else if (mode == 1) {
          const int bb = m >> 11;          // m / S_DIM
          const int s  = m & (S_DIM - 1);
          ((bf16_t*)Cout)[((size_t)bb * D_DIM + n) * S_DIM + s] = (bf16_t)v;
        } else {
          ((float*)Cout)[(size_t)m * N + n] = v;
        }
      }
  }
}

// ---------------------------------------------------------------------------
// Attention with softmax over the HEAD axis (faithful to reference).
// Block = 512 thr = 16 waves; wave h handles head h. One 16-query tile per
// block, streaming 32 keys per iteration. For each (q,k) the 16 head scores
// are exp'd and sum-reduced across waves via LDS ds_add_f32; no key-axis
// running state is needed since normalization is per-(q,k) across heads.
// ---------------------------------------------------------------------------
__global__ __launch_bounds__(512)
void head_softmax_attention_kernel(const bf16_t* __restrict__ Qb,
                                   const bf16_t* __restrict__ Kb,
                                   const bf16_t* __restrict__ Vt,
                                   bf16_t* __restrict__ Ob) {
  __shared__ float sums[16 * 32];                        // [q in tile][k in tile]
  __shared__ __align__(16) bf16_t attnT[H_DIM][16 * 32]; // per-wave attn tile

  const int tid  = threadIdx.x;
  const int lane = tid & 31;
  const int h    = tid >> 5;             // wave id == head
  const int b    = blockIdx.x >> 7;      // / (S/16)
  const int qt   = blockIdx.x & 127;
  const int qbase = qt * 16;
  const int lrow = lane & 15;
  const int lhi  = lane >> 4;

  // Q tile for this head: 16x64 as two 16x32 A-operands (resident whole loop)
  const bf16_t* qrow = Qb + ((size_t)(b * S_DIM + qbase + lrow)) * D_DIM + h * HD_DIM;
  V16 qa[2];
#pragma unroll
  for (int kh = 0; kh < 2; ++kh) {
    const int c = lhi * 8 + kh * 32;
    qa[kh].h[0] = load8bf(qrow + c);
    qa[kh].h[1] = load8bf(qrow + c + 16);
  }

  v8f acc[4] = {};                       // out tile 16x64 (4 N-tiles), f32
  const float scale = 0.125f;            // 1/sqrt(HD)

  for (int kb = 0; kb < S_DIM; kb += 32) {
    sums[tid] = 0.0f;                    // 512 threads zero 16*32 entries
    __syncthreads();

    // scores: two 16x16 C tiles (keys kb..kb+16, kb+16..kb+32)
    float e[2][8];
#pragma unroll
    for (int t = 0; t < 2; ++t) {
      const bf16_t* krow =
          Kb + ((size_t)(b * S_DIM + kb + t * 16 + lrow)) * D_DIM + h * HD_DIM + lhi * 8;
      V16 b0, b1;
      b0.h[0] = load8bf(krow);             // features 0..31
      b0.h[1] = load8bf(krow + 16);
      b1.h[0] = load8bf(krow + 32);        // features 32..63
      b1.h[1] = load8bf(krow + 48);
      v8f c = {};
      c = __builtin_amdgcn_wmma_f32_16x16x32_bf16(false, qa[0].v, false, b0.v,
                                                  (short)0, c, false, false);
      c = __builtin_amdgcn_wmma_f32_16x16x32_bf16(false, qa[1].v, false, b1.v,
                                                  (short)0, c, false, false);
#pragma unroll
      for (int r = 0; r < 8; ++r) {
        const float ev = __expf(c[r] * scale);
        e[t][r] = ev;
        atomicAdd(&sums[(r + 8 * lhi) * 32 + t * 16 + lrow], ev);  // ds_add_f32
      }
    }
    __syncthreads();

    // normalize across heads; stash attn (bf16) in this wave's LDS tile to
    // convert C-layout -> A-layout (16x32). Wave-private; backend inserts
    // the s_wait_dscnt for the RAW below.
#pragma unroll
    for (int t = 0; t < 2; ++t)
#pragma unroll
      for (int r = 0; r < 8; ++r) {
        const int m = r + 8 * lhi;
        const int n = t * 16 + lrow;
        const float av = e[t][r] * __builtin_amdgcn_rcpf(sums[m * 32 + n]);
        attnT[h][m * 32 + n] = (bf16_t)av;
      }

    V16 aa;
    {
      const bf16_t* ar = &attnT[h][lrow * 32 + lhi * 8];
      aa.h[0] = load8bf(ar);
      aa.h[1] = load8bf(ar + 16);
    }

    // attn(16x32) @ V(32x64): Vt[b][d][s] makes B-operand contiguous per lane
#pragma unroll
    for (int dt = 0; dt < 4; ++dt) {
      const bf16_t* vr =
          Vt + ((size_t)(b * D_DIM) + h * HD_DIM + dt * 16 + lrow) * S_DIM + kb + lhi * 8;
      V16 bv;
      bv.h[0] = load8bf(vr);
      bv.h[1] = load8bf(vr + 16);
      acc[dt] = __builtin_amdgcn_wmma_f32_16x16x32_bf16(false, aa.v, false, bv.v,
                                                        (short)0, acc[dt], false, false);
    }
    __syncthreads();   // all waves done reading sums before next re-zero
  }

  // store attention output tile (bf16, row-major [b,s,D]) for final projection
#pragma unroll
  for (int dt = 0; dt < 4; ++dt)
#pragma unroll
    for (int r = 0; r < 8; ++r) {
      const int m = qbase + r + 8 * lhi;
      const int n = h * HD_DIM + dt * 16 + lrow;
      Ob[((size_t)(b * S_DIM + m)) * D_DIM + n] = (bf16_t)acc[dt][r];
    }
}

// ---------------------------------------------------------------------------
extern "C" void kernel_launch(void* const* d_in, const int* in_sizes, int n_in,
                              void* d_out, int out_size, void* d_ws, size_t ws_size,
                              hipStream_t stream) {
  (void)in_sizes; (void)n_in; (void)out_size; (void)ws_size;

  const float* x  = (const float*)d_in[0];
  const float* Wq = (const float*)d_in[1];
  const float* bq = (const float*)d_in[2];
  const float* Wk = (const float*)d_in[3];
  const float* bk = (const float*)d_in[4];
  const float* Wv = (const float*)d_in[5];
  const float* bv = (const float*)d_in[6];
  const float* Wo = (const float*)d_in[7];
  const float* bo = (const float*)d_in[8];

  // workspace carve-out (~48 MB total)
  char* ws = (char*)d_ws;
  size_t off = 0;
  auto carve = [&](size_t bytes) -> void* {
    off = (off + 255) & ~(size_t)255;
    void* p = ws + off;
    off += bytes;
    return p;
  };
  const size_t nX = (size_t)M_DIM * D_DIM;   // 4096*1024
  const size_t nW = (size_t)D_DIM * D_DIM;   // 1024*1024
  bf16_t* Xb  = (bf16_t*)carve(nX * 2);
  bf16_t* Wqb = (bf16_t*)carve(nW * 2);
  bf16_t* Wkb = (bf16_t*)carve(nW * 2);
  bf16_t* Wvb = (bf16_t*)carve(nW * 2);
  bf16_t* Wob = (bf16_t*)carve(nW * 2);
  bf16_t* Qb  = (bf16_t*)carve(nX * 2);
  bf16_t* Kb  = (bf16_t*)carve(nX * 2);
  bf16_t* Vt  = (bf16_t*)carve(nX * 2);      // [b][d][s]
  bf16_t* Ob  = (bf16_t*)carve(nX * 2);

  // 1) convert inputs to bf16
  cvt_f32_bf16_kernel<<<2048, 256, 0, stream>>>(x,  Xb,  (int)nX);
  cvt_f32_bf16_kernel<<<1024, 256, 0, stream>>>(Wq, Wqb, (int)nW);
  cvt_f32_bf16_kernel<<<1024, 256, 0, stream>>>(Wk, Wkb, (int)nW);
  cvt_f32_bf16_kernel<<<1024, 256, 0, stream>>>(Wv, Wvb, (int)nW);
  cvt_f32_bf16_kernel<<<1024, 256, 0, stream>>>(Wo, Wob, (int)nW);

  // 2) projections
  dim3 ggrid(M_DIM / 64, D_DIM / 256);
  gemm_xwT_kernel<<<ggrid, 256, 0, stream>>>(Xb, Wqb, bq, Qb, M_DIM, D_DIM, D_DIM, 0);
  gemm_xwT_kernel<<<ggrid, 256, 0, stream>>>(Xb, Wkb, bk, Kb, M_DIM, D_DIM, D_DIM, 0);
  gemm_xwT_kernel<<<ggrid, 256, 0, stream>>>(Xb, Wvb, bv, Vt, M_DIM, D_DIM, D_DIM, 1);

  // 3) attention with head-axis softmax
  head_softmax_attention_kernel<<<B_DIM * (S_DIM / 16), 512, 0, stream>>>(Qb, Kb, Vt, Ob);

  // 4) output projection (f32 result)
  gemm_xwT_kernel<<<ggrid, 256, 0, stream>>>(Ob, Wob, bo, (float*)d_out,
                                             M_DIM, D_DIM, D_DIM, 2);
}
